// MyParticleNetwork_73126113182389
// MI455X (gfx1250) — matrix operand order repaired and checked by reading, not compiled, not run on hardware
//
#include <hip/hip_runtime.h>
#include <hip/hip_bf16.h>
#include <math.h>

// ---------------------------------------------------------------------------
// Constants from the reference
// ---------------------------------------------------------------------------
#define DT_STEP      (1.0f / 50.0f)
#define RADIUS_F     (0.5f * 1.5f * 6.0f * 0.025f)   // 0.1125
#define KSZ          4
#define K3CELLS      64
#define QB           16          // queries per workgroup (16 wave32s)
#define CIN_PASS_MAX 32          // max input channels per cconv pass
#define KTOT_MAX     (K3CELLS * CIN_PASS_MAX)  // 2048
#define GPAD         8           // +32B row padding to break LDS bank conflicts
#define NT_MAX       4           // max N tiles (cout<=64)

typedef __bf16 v16bf __attribute__((ext_vector_type(16)));
typedef float  v8f   __attribute__((ext_vector_type(8)));

__device__ __forceinline__ float signf(float x) {
    return (x > 0.f) ? 1.f : ((x < 0.f) ? -1.f : 0.f);
}

// A-fragment: 16 bf16 for v_wmma_f32_16x16x32_bf16, converted from fp32 LDS.
// gp already includes row*stride + koff; lane j: K = kc*32 + koff + j / +16+j.
__device__ __forceinline__ v16bf load_a_frag(const float* __restrict__ gp) {
    v16bf a;
    #pragma unroll
    for (int j = 0; j < 8; ++j) a[j]     = (__bf16)gp[j];
    #pragma unroll
    for (int j = 0; j < 8; ++j) a[8 + j] = (__bf16)gp[16 + j];
    return a;
}

// ---------------------------------------------------------------------------
// Integrator prep: pos2 = pos + dt*(vel2+vel)/2, fl = [1, vel2]
// ---------------------------------------------------------------------------
__global__ void pn_prep(const float* __restrict__ pos, const float* __restrict__ vel,
                        float* __restrict__ pos2, float* __restrict__ fl, int n) {
    int i = blockIdx.x * blockDim.x + threadIdx.x;
    if (i >= n) return;
    float vx = vel[i*3+0], vy = vel[i*3+1], vz = vel[i*3+2];
    float v2x = vx, v2y = vy - DT_STEP * 9.81f, v2z = vz;
    pos2[i*3+0] = pos[i*3+0] + DT_STEP * (v2x + vx) * 0.5f;
    pos2[i*3+1] = pos[i*3+1] + DT_STEP * (v2y + vy) * 0.5f;
    pos2[i*3+2] = pos[i*3+2] + DT_STEP * (v2z + vz) * 0.5f;
    fl[i*4+0] = 1.f; fl[i*4+1] = v2x; fl[i*4+2] = v2y; fl[i*4+3] = v2z;
}

// ---------------------------------------------------------------------------
// CSR row offsets: rs[q] = lower_bound(qidx, q); qidx is sorted ascending.
// ---------------------------------------------------------------------------
__global__ void pn_csr(const int* __restrict__ qidx, int E, int n, int* __restrict__ rs) {
    int q = blockIdx.x * blockDim.x + threadIdx.x;
    if (q > n) return;
    int lo = 0, hi = E;
    while (lo < hi) { int mid = (lo + hi) >> 1; if (qidx[mid] < q) lo = mid + 1; else hi = mid; }
    rs[q] = lo;
}

// ---------------------------------------------------------------------------
// Per-edge geometry: window * trilinear corner weights (8) + cell ids (8).
// Shared by every cconv over the same edge list.
// ---------------------------------------------------------------------------
__global__ void pn_edge_geom(const int* __restrict__ q_idx, const int* __restrict__ s_idx,
                             const float* __restrict__ pos_q, const float* __restrict__ pos_s,
                             int E, float* __restrict__ wgt8, unsigned char* __restrict__ cell8) {
    int e = blockIdx.x * blockDim.x + threadIdx.x;
    if (e >= E) return;
    int q = q_idx[e], s = s_idx[e];
    const float invR = 1.0f / RADIUS_F;
    float x = (pos_s[s*3+0] - pos_q[q*3+0]) * invR;
    float y = (pos_s[s*3+1] - pos_q[q*3+1]) * invR;
    float z = (pos_s[s*3+2] - pos_q[q*3+2]) * invR;
    float r2 = x*x + y*y + z*z;
    float w1 = 1.f - r2;
    float win = fminf(fmaxf(w1*w1*w1, 0.f), 1.f);

    // ball_to_cube
    const float eps = 1e-12f;
    float nrm = sqrtf(r2 + eps);
    bool  cap = (1.25f * z * z) > (x*x + y*y);
    float s_cap  = sqrtf(3.f * nrm / (nrm + fabsf(z) + eps));
    float s_side = nrm / sqrtf(x*x + y*y + eps);
    float sc = cap ? s_cap : s_side;
    float xc = x * sc, yc = y * sc;
    float zc = cap ? (signf(z) * nrm) : (1.5f * z);
    if (!(r2 > eps)) { xc = 0.f; yc = 0.f; zc = 0.f; }
    float rxy  = sqrtf(xc*xc + yc*yc + eps);
    bool  xbig = fabsf(yc) <= fabsf(xc);
    float dx = (fabsf(xc) > eps) ? xc : 1.f;
    float dy = (fabsf(yc) > eps) ? yc : 1.f;
    const float c4pi = 4.f / 3.14159265358979f;
    float tx = signf(xc) * rxy, ty = signf(yc) * rxy;
    float xq = xbig ? tx : (ty * c4pi * atanf(xc / dy));
    float yq = xbig ? (tx * c4pi * atanf(yc / dx)) : ty;
    if (!(xc*xc + yc*yc > eps)) { xq = 0.f; yq = 0.f; }

    float g[3]; g[0] = (xq + 1.f) * 1.5f; g[1] = (yq + 1.f) * 1.5f; g[2] = (zc + 1.f) * 1.5f;
    float fr[3]; int g0[3];
    #pragma unroll
    for (int d = 0; d < 3; ++d) { float gf = floorf(g[d]); fr[d] = g[d] - gf; g0[d] = (int)gf; }
    #pragma unroll
    for (int c = 0; c < 8; ++c) {
        int b0 = (c >> 2) & 1, b1 = (c >> 1) & 1, b2 = c & 1;
        int i0 = min(max(g0[0] + b0, 0), KSZ - 1);
        int i1 = min(max(g0[1] + b1, 0), KSZ - 1);
        int i2 = min(max(g0[2] + b2, 0), KSZ - 1);
        float w = (b0 ? fr[0] : 1.f - fr[0]) * (b1 ? fr[1] : 1.f - fr[1]) * (b2 ? fr[2] : 1.f - fr[2]);
        wgt8[(size_t)e*8 + c]  = win * w;
        cell8[(size_t)e*8 + c] = (unsigned char)((i0 * KSZ + i1) * KSZ + i2);
    }
}

// ---------------------------------------------------------------------------
// Repack conv weights [64][cin_tot][cout] fp32 -> bf16 in the exact per-lane
// layout of the v_wmma_f32_16x16x32_bf16 B operand (32x16 per tile):
//   Wp[((nt*kchunks + kc)*32 + lane)*16 + j]
//   = W[k = kc*32 + (j<8 ? (lane>=16)*8 + j : 16 + (lane>=16)*8 + j-8)][col = nt*16 + (lane&15)], zero-padded.
// ---------------------------------------------------------------------------
__global__ void pn_repack_w(const float* __restrict__ W, __bf16* __restrict__ Wp,
                            int cin_tot, int ci_off, int cin_pass, int cout,
                            int ntiles, int kchunks) {
    int total = ntiles * kchunks * 512;
    int t = blockIdx.x * blockDim.x + threadIdx.x;
    if (t >= total) return;
    int j    = t & 15;
    int lane = (t >> 4) & 31;
    int kc   = (t >> 9) % kchunks;
    int nt   = t / (kchunks * 512);
    int k    = kc * 32 + ((lane >= 16) ? 8 : 0) + ((j < 8) ? j : 8 + j);
    int col  = nt * 16 + (lane & 15);
    int cell = k / cin_pass;
    int ci   = k - cell * cin_pass;
    float v = 0.f;
    if (col < cout) v = W[((size_t)(cell * cin_tot + ci_off + ci)) * cout + col];
    Wp[t] = (__bf16)v;
}

// ---------------------------------------------------------------------------
// Dense: out[q, co_off+co] = h @ wd + bd (+ resid)
// ---------------------------------------------------------------------------
__global__ void pn_dense(const float* __restrict__ h, int ldh, int cin,
                         const float* __restrict__ wd, const float* __restrict__ bd,
                         int cout, float* __restrict__ out, int ldo, int co_off,
                         const float* __restrict__ resid, int ldr, int n) {
    int t = blockIdx.x * blockDim.x + threadIdx.x;
    if (t >= n * cout) return;
    int q = t / cout, co = t - q * cout;
    float acc = bd[co];
    for (int ci = 0; ci < cin; ++ci) acc += h[(size_t)q * ldh + ci] * wd[(size_t)ci * cout + co];
    if (resid) acc += resid[(size_t)q * ldr + co];
    out[(size_t)q * ldo + co_off + co] = acc;
}

__global__ void pn_relu(const float* __restrict__ x, int ldx,
                        float* __restrict__ h, int ldh, int n, int ch) {
    int t = blockIdx.x * blockDim.x + threadIdx.x;
    if (t >= n * ch) return;
    int q = t / ch, c = t - q * ch;
    h[(size_t)q * ldh + c] = fmaxf(x[(size_t)q * ldx + c], 0.f);
}

// ---------------------------------------------------------------------------
// Fused cconv: scatter into LDS G tile (padded stride, bank-conflict free GEMM
// reads), then WMMA GEMM G[16,Ktot] x W[Ktot,cout], K loop unrolled x2,
// cross-wave partial-C reduction via ds_add_f32.
// ---------------------------------------------------------------------------
__global__ void __launch_bounds__(512, 1)
pn_cconv(const float* __restrict__ feat, int ld_feat, int ci_off, int cin_pass,
         const int* __restrict__ s_idx, const int* __restrict__ rs,
         const float* __restrict__ wgt8, const unsigned char* __restrict__ cell8,
         const __bf16* __restrict__ Wp, int cout, int ntiles, int kchunks,
         float* __restrict__ out, int ldo, int co_off, int n) {
    __shared__ float Gs[QB * (KTOT_MAX + GPAD)];   // ~128.5 KB fp32 scatter tile
    __shared__ float Cbuf[NT_MAX * 256];           // 4 KB partial-C reduction

    const int tid  = threadIdx.x;
    const int lane = tid & 31;
    const int wv   = tid >> 5;
    const int Ktot = K3CELLS * cin_pass;
    const int gstr = Ktot + GPAD;                  // padded row stride (breaks bank aliasing)
    const int q0   = blockIdx.x * QB;

    for (int i = tid; i < QB * gstr; i += 512) Gs[i] = 0.f;
    for (int i = tid; i < NT_MAX * 256; i += 512) Cbuf[i] = 0.f;
    __syncthreads();

    // ---- scatter phase: wave wv handles query q0+wv ----
    int q = q0 + wv;
    if (q < n) {
        int e0 = rs[q], e1 = rs[q + 1];
        bool act = lane < cin_pass;
        float* Gw = &Gs[wv * gstr];
        for (int e = e0; e < e1; ++e) {
            int s = s_idx[e];
            float fv = act ? feat[(size_t)s * ld_feat + ci_off + lane] : 0.f;
            const float4* wp4 = (const float4*)(wgt8 + (size_t)e * 8);
            float4 wa = wp4[0], wb = wp4[1];
            unsigned long long cl = *(const unsigned long long*)(cell8 + (size_t)e * 8);
            float w8[8] = {wa.x, wa.y, wa.z, wa.w, wb.x, wb.y, wb.z, wb.w};
            if (act) {
                #pragma unroll
                for (int k = 0; k < 8; ++k) {
                    int c = (int)((cl >> (8 * k)) & 63ull);
                    Gw[c * cin_pass + lane] += w8[k] * fv;  // race-free: lane owns channel
                }
            }
        }
    }
    __syncthreads();

    // ---- GEMM phase: wave wv -> N-tile (wv % ntiles), K-chunks strided ----
    const int nt     = wv % ntiles;
    const int kstart = wv / ntiles;
    const int kstep  = 16 / ntiles;     // ntiles in {1,2,4}
    v8f acc = {0.f, 0.f, 0.f, 0.f, 0.f, 0.f, 0.f, 0.f};
    const int row  = lane & 15;
    const int koff = (lane >> 4) * 8;
    const float*  abase = &Gs[row * gstr + koff];
    const __bf16* bptr  = Wp + ((size_t)(nt * kchunks + kstart) * 32 + lane) * 16;
    const size_t  bstep = (size_t)kstep * 512;     // elements per K-chunk stride

    int kc = kstart;
    for (; kc + kstep < kchunks; kc += 2 * kstep) {
        v16bf a0 = load_a_frag(abase + kc * 32);
        v16bf b0 = *(const v16bf*)bptr;
        v16bf a1 = load_a_frag(abase + (kc + kstep) * 32);
        v16bf b1 = *(const v16bf*)(bptr + bstep);
        acc = __builtin_amdgcn_wmma_f32_16x16x32_bf16(false, a0, false, b0,
                                                      (short)0, acc, false, false);
        acc = __builtin_amdgcn_wmma_f32_16x16x32_bf16(false, a1, false, b1,
                                                      (short)0, acc, false, false);
        bptr += 2 * bstep;
    }
    for (; kc < kchunks; kc += kstep) {
        v16bf a0 = load_a_frag(abase + kc * 32);
        v16bf b0 = *(const v16bf*)bptr;
        acc = __builtin_amdgcn_wmma_f32_16x16x32_bf16(false, a0, false, b0,
                                                      (short)0, acc, false, false);
        bptr += bstep;
    }

    const int rbase = (lane < 16) ? 0 : 8;
    const int colc  = lane & 15;
    #pragma unroll
    for (int r = 0; r < 8; ++r)
        atomicAdd(&Cbuf[nt * 256 + (rbase + r) * 16 + colc], acc[r]);
    __syncthreads();

    // ---- accumulate into output ----
    for (int t = tid; t < ntiles * 256; t += 512) {
        int ntile = t >> 8, m = (t >> 4) & 15, cc = t & 15;
        int col = ntile * 16 + cc, qq = q0 + m;
        if (qq < n && col < cout)
            out[(size_t)qq * ldo + co_off + col] += Cbuf[t];
    }
}

// ---------------------------------------------------------------------------
// Finish: pos_new = pos2 + x/128 ; vel_new = (pos_new - pos)/dt
// ---------------------------------------------------------------------------
__global__ void pn_finish(const float* __restrict__ pos, const float* __restrict__ pos2,
                          const float* __restrict__ x, float* __restrict__ out, int n) {
    int i = blockIdx.x * blockDim.x + threadIdx.x;
    if (i >= n) return;
    #pragma unroll
    for (int d = 0; d < 3; ++d) {
        float pn = pos2[i*3+d] + x[(size_t)i * 96 + d] * (1.f / 128.f);
        out[i*3+d] = pn;
        out[(size_t)n * 3 + i*3+d] = (pn - pos[i*3+d]) * (1.f / DT_STEP);
    }
}

// ---------------------------------------------------------------------------
// Host orchestration
// ---------------------------------------------------------------------------
extern "C" void kernel_launch(void* const* d_in, const int* in_sizes, int n_in,
                              void* d_out, int out_size, void* d_ws, size_t ws_size,
                              hipStream_t stream) {
    (void)n_in; (void)out_size; (void)ws_size;
    const int n    = in_sizes[0] / 3;
    const int Eff  = in_sizes[4];
    const int Efo  = in_sizes[6];

    const float* pos   = (const float*)d_in[0];
    const float* vel   = (const float*)d_in[1];
    const float* box   = (const float*)d_in[2];
    const float* bfeat = (const float*)d_in[3];
    const int*   ffq   = (const int*)d_in[4];
    const int*   ffs   = (const int*)d_in[5];
    const int*   foq   = (const int*)d_in[6];
    const int*   fos   = (const int*)d_in[7];
    const float* w_cf0 = (const float*)d_in[8];
    const float* w_co0 = (const float*)d_in[9];
    const float* w_d0  = (const float*)d_in[10];
    const float* b_d0  = (const float*)d_in[11];
    const float* w_c1  = (const float*)d_in[12];
    const float* w_d1  = (const float*)d_in[13];
    const float* b_d1  = (const float*)d_in[14];
    const float* w_c2  = (const float*)d_in[15];
    const float* w_d2  = (const float*)d_in[16];
    const float* b_d2  = (const float*)d_in[17];
    const float* w_c3  = (const float*)d_in[18];
    const float* w_d3  = (const float*)d_in[19];
    const float* b_d3  = (const float*)d_in[20];

    // workspace carve-out
    char* base = (char*)d_ws; size_t off = 0;
    auto carve = [&](size_t bytes) -> void* {
        void* r = base + off; off = (off + bytes + 255) & ~(size_t)255; return r;
    };
    float*         pos2    = (float*)carve((size_t)n * 3 * 4);
    float*         fl      = (float*)carve((size_t)n * 4 * 4);
    float*         Abuf    = (float*)carve((size_t)n * 96 * 4);
    float*         Bbuf    = (float*)carve((size_t)n * 96 * 4);
    float*         Hbuf    = (float*)carve((size_t)n * 96 * 4);
    int*           rs_ff   = (int*)carve((size_t)(n + 1) * 4);
    int*           rs_fo   = (int*)carve((size_t)(n + 1) * 4);
    float*         wgt_ff  = (float*)carve((size_t)Eff * 8 * 4);
    unsigned char* cell_ff = (unsigned char*)carve((size_t)Eff * 8);
    float*         wgt_fo  = (float*)carve((size_t)Efo * 8 * 4);
    unsigned char* cell_fo = (unsigned char*)carve((size_t)Efo * 8);
    __bf16*        Wp      = (__bf16*)carve((size_t)KTOT_MAX * 64 * 2);

    const int TB = 256;
    auto blks = [&](long long work) { return (int)((work + TB - 1) / TB); };

    pn_prep<<<blks(n), TB, 0, stream>>>(pos, vel, pos2, fl, n);
    pn_csr<<<blks(n + 1), TB, 0, stream>>>(ffq, Eff, n, rs_ff);
    pn_csr<<<blks(n + 1), TB, 0, stream>>>(foq, Efo, n, rs_fo);
    if (Eff > 0) pn_edge_geom<<<blks(Eff), TB, 0, stream>>>(ffq, ffs, pos2, pos2, Eff, wgt_ff, cell_ff);
    if (Efo > 0) pn_edge_geom<<<blks(Efo), TB, 0, stream>>>(foq, fos, pos2, box, Efo, wgt_fo, cell_fo);
    hipMemsetAsync(Abuf, 0, (size_t)n * 96 * 4, stream);

    auto run_conv = [&](const float* W, int cin_tot, int ci_off, int cin_pass, int cout,
                        const float* feat, int ld_feat,
                        const int* sidx, const int* rs, const float* wgt,
                        const unsigned char* cells, float* outp, int co_off) {
        int ntiles  = (cout + 15) / 16;
        int kchunks = (K3CELLS * cin_pass) / 32;
        int total   = ntiles * kchunks * 512;
        pn_repack_w<<<blks(total), TB, 0, stream>>>(W, Wp, cin_tot, ci_off, cin_pass, cout, ntiles, kchunks);
        pn_cconv<<<(n + QB - 1) / QB, 512, 0, stream>>>(feat, ld_feat, ci_off, cin_pass,
                                                        sidx, rs, wgt, cells, Wp,
                                                        cout, ntiles, kchunks, outp, 96, co_off, n);
    };

    // ---- layer 0: x = [a_co(32) | a_cf(32) | a_df(32)] in Abuf (ld 96) ----
    run_conv(w_co0, 3, 0, 3, 32, bfeat, 3, fos, rs_fo, wgt_fo, cell_fo, Abuf, 0);
    run_conv(w_cf0, 4, 0, 4, 32, fl,    4, ffs, rs_ff, wgt_ff, cell_ff, Abuf, 32);
    pn_dense<<<blks((long long)n * 32), TB, 0, stream>>>(fl, 4, 4, w_d0, b_d0, 32, Abuf, 96, 64, nullptr, 0, n);

    // ---- layer 1: 96 -> 64 (no residual), output Bbuf ----
    pn_relu<<<blks((long long)n * 96), TB, 0, stream>>>(Abuf, 96, Hbuf, 96, n, 96);
    pn_dense<<<blks((long long)n * 64), TB, 0, stream>>>(Hbuf, 96, 96, w_d1, b_d1, 64, Bbuf, 96, 0, nullptr, 0, n);
    for (int p = 0; p < 3; ++p)
        run_conv(w_c1, 96, 32 * p, 32, 64, Hbuf, 96, ffs, rs_ff, wgt_ff, cell_ff, Bbuf, 0);

    // ---- layer 2: 64 -> 64 (residual), output Abuf ----
    pn_relu<<<blks((long long)n * 64), TB, 0, stream>>>(Bbuf, 96, Hbuf, 96, n, 64);
    pn_dense<<<blks((long long)n * 64), TB, 0, stream>>>(Hbuf, 96, 64, w_d2, b_d2, 64, Abuf, 96, 0, Bbuf, 96, n);
    for (int p = 0; p < 2; ++p)
        run_conv(w_c2, 64, 32 * p, 32, 64, Hbuf, 96, ffs, rs_ff, wgt_ff, cell_ff, Abuf, 0);

    // ---- layer 3: 64 -> 3 (no residual), output Bbuf ----
    pn_relu<<<blks((long long)n * 64), TB, 0, stream>>>(Abuf, 96, Hbuf, 96, n, 64);
    pn_dense<<<blks((long long)n * 3), TB, 0, stream>>>(Hbuf, 96, 64, w_d3, b_d3, 3, Bbuf, 96, 0, nullptr, 0, n);
    for (int p = 0; p < 2; ++p)
        run_conv(w_c3, 64, 32 * p, 32, 3, Hbuf, 96, ffs, rs_ff, wgt_ff, cell_ff, Bbuf, 0);

    pn_finish<<<blks(n), TB, 0, stream>>>(pos, pos2, Bbuf, (float*)d_out, n);
}
